// RGAKAN_76596446757518
// MI455X (gfx1250) — matrix operand order repaired and checked by reading, not compiled, not run on hardware
//
#include <hip/hip_runtime.h>
#include <hip/hip_bf16.h>
#include <stdint.h>

typedef __attribute__((ext_vector_type(16))) __bf16 v16bf;
typedef __attribute__((ext_vector_type(8)))  float  v8f;

#define BATCH   32768
#define NHID    256
#define DDEG    8
#define KDIM    2048              // NHID * DDEG
#define WELEMS  (KDIM * NHID)     // 524288 elems per weight tensor
#define MTILE   64                // rows per workgroup
#define LDSSTRIDE 66              // padded m-stride for tanh tile (bf16)
#define TSH_BYTES (NHID * LDSSTRIDE * 2)   // 33792
#define BSTAGE_BYTES 16384                  // one K-tile weight slice (32x256 bf16)
#define NBUF 3                              // async pipeline depth (2 tiles ahead)
#define SMEM_TOTAL (TSH_BYTES + NBUF * BSTAGE_BYTES)  // 82944 -> 3 WGs/WGP

// ---------------------------------------------------------------------------
// Repack one weight tensor C[o,i,d] (f32, o=N, K=i*8+d) into bf16 fragments
// for the v_wmma_f32_16x16x32_bf16 B operand, in contiguous k order (as kan_gemm's A):
//   fragment (kt, nt): lane L holds 16 bf16, element e -> N = nt*16 + L%16,
//   K = kt*32 + (L/16)*16 + e.  Lane data is 32B contiguous; a whole K-tile
//   slice (16 nt * 32 lanes * 32B = 16KB) is contiguous for async staging.
// ---------------------------------------------------------------------------
__global__ __launch_bounds__(256) void prep_w(const float* __restrict__ W,
                                              unsigned short* __restrict__ Wf) {
    int tid  = blockIdx.x * blockDim.x + threadIdx.x;   // 0..32767
    int lane = tid & 31;
    int frag = tid >> 5;                                // 0..1023
    int nt   = frag & 15;
    int kt   = frag >> 4;                               // 0..63
    int n     = nt * 16 + (lane & 15);
    int kbase = kt * 32 + (lane >> 4) * 16;
    v16bf o;
#pragma unroll
    for (int e = 0; e < 16; ++e)
        o[e] = (__bf16)W[(size_t)n * KDIM + kbase + e];
    *(v16bf*)((__bf16*)Wf + (size_t)tid * 16) = o;
}

// ---------------------------------------------------------------------------
// RFF embedding: h[b, 0:128] = cos(x@B), h[b, 128:256] = sin(x@B)
// ---------------------------------------------------------------------------
__global__ __launch_bounds__(256) void rff_kernel(const float* __restrict__ x,
                                                  const float* __restrict__ B,
                                                  float* __restrict__ h) {
    int t = blockIdx.x * blockDim.x + threadIdx.x;      // 0..32768*128-1
    int b = t >> 7;
    int j = t & 127;
    float z = x[b * 3 + 0] * B[j] + x[b * 3 + 1] * B[128 + j] +
              x[b * 3 + 2] * B[256 + j];
    float s, c;
    sincosf(z, &s, &c);
    h[(size_t)b * NHID + j]       = c;
    h[(size_t)b * NHID + 128 + j] = s;
}

// ---------------------------------------------------------------------------
// Fused Chebyshev-KAN GEMM:  out = ChebBasis(tanh(act)) @ W^T  (+ epilogue)
//   mode 0: out = gemm
//   mode 1: t = gemm; w = t*u + (1-t)*v; out = s*w + (1-s)*identity
// Block = 128 threads (4 waves), M-tile = 64, full N = 256 per block.
// Weight K-slices: triple-buffered in LDS via global_load_async_to_lds_b128
// (ASYNCcnt), issued TWO K-tiles ahead.  B fragments: two live register sets
// (B0/B1) interleaved so every ds_load has a full WMMA of latency cover.
// ---------------------------------------------------------------------------
__global__ __launch_bounds__(128) void kan_gemm(const float* __restrict__ act,
                                                const unsigned short* __restrict__ wfu,
                                                float* __restrict__ outp,
                                                const float* __restrict__ up,
                                                const float* __restrict__ vp,
                                                const float* __restrict__ idp,
                                                const float* __restrict__ scal,
                                                int mode) {
    extern __shared__ char smem[];
    __bf16* tsh = (__bf16*)smem;                // [256][LDSSTRIDE] transposed tanh (bf16)
    char*   bst = smem + TSH_BYTES;             // 3 x 16KB weight staging
    const int tid = threadIdx.x;
    const int m0  = blockIdx.x * MTILE;

    const uint32_t bst_lds = (uint32_t)(uintptr_t)bst;   // LDS byte offset (addr[31:0])
    const char* wbytes = (const char*)wfu;

    // --- async prefetch of K-tile kt into staging buffer buf (8 x b128/thread)
    auto prefetchB = [&](int kt, int buf) {
        uint64_t g = (uint64_t)(uintptr_t)(wbytes + (size_t)kt * BSTAGE_BYTES) +
                     (uint32_t)(tid * 16);
        uint32_t l = bst_lds + (uint32_t)(buf * BSTAGE_BYTES + tid * 16);
#pragma unroll
        for (int j = 0; j < 8; ++j) {           // 128 thr * 16B = 2KB per j
            asm volatile("global_load_async_to_lds_b128 %0, %1, off"
                         :: "v"(l), "v"(g) : "memory");
            g += 2048; l += 2048;
        }
    };

    prefetchB(0, 0);                            // overlap with tanh staging below
    prefetchB(1, 1);

    // Stage tanh(act) tile, transposed: tsh[i*66 + m_local] (bf16)
    for (int k = tid; k < MTILE * NHID; k += 128) {
        int m = k >> 8;
        int c = k & 255;
        tsh[c * LDSSTRIDE + m] = (__bf16)tanhf(act[(size_t)(m0 + m) * NHID + c]);
    }

    const int lane = tid & 31;
    const int wave = tid >> 5;
    const int half = lane >> 4;
    const int ml   = wave * 16 + (lane & 15);   // local row 0..63

    v8f acc[16];
    v8f zero = {};
#pragma unroll
    for (int nt = 0; nt < 16; ++nt) acc[nt] = zero;

#pragma unroll 1
    for (int kt = 0; kt < 64; ++kt) {
        // In-order async completion: <=8 outstanding means this tile's 8 copies
        // (the oldest) are done while the next tile's 8 remain in flight.
        if (kt < 63)
            asm volatile("s_wait_asynccnt 0x8" ::: "memory");
        else
            asm volatile("s_wait_asynccnt 0x0" ::: "memory");
        __syncthreads();                                 // everyone's copies visible
        if (kt < 62) prefetchB(kt + 2, (kt + 2) % NBUF); // overlaps the WMMAs below

        const __bf16* bl = (const __bf16*)(bst + (kt % NBUF) * BSTAGE_BYTES) + lane * 16;
        v16bf B0 = *(const v16bf*)(bl);                  // latency hides under A-build
        v16bf B1 = *(const v16bf*)(bl + 512);

        // A fragment, contiguous k order (K = kt*32 + 16*half + e, as in prep_w):
        //   element e<8 : i = kt*4 + 2*half,     d = e
        //   element e>=8: i = kt*4 + 2*half + 1, d = e-8
        int   i0 = kt * 4 + 2 * half;
        float ta = (float)tsh[i0 * LDSSTRIDE + ml];
        float tb = (float)tsh[(i0 + 1) * LDSSTRIDE + ml];
        v16bf A;
        A[0] = (__bf16)1.0f; A[1] = (__bf16)ta;
        A[8] = (__bf16)1.0f; A[9] = (__bf16)tb;
        float pa0 = 1.0f, pa1 = ta, pb0 = 1.0f, pb1 = tb;
#pragma unroll
        for (int d = 2; d < 8; ++d) {           // Chebyshev recurrence
            float na = 2.0f * ta * pa1 - pa0; pa0 = pa1; pa1 = na;
            float nb = 2.0f * tb * pb1 - pb0; pb0 = pb1; pb1 = nb;
            A[d]     = (__bf16)na;
            A[8 + d] = (__bf16)nb;
        }

        // Two interleaved chains: B0 and B1 are both live across every WMMA,
        // so each reload overlaps the other chain's WMMA (wait dscnt<=2).
#pragma unroll
        for (int nt = 0; nt < 16; nt += 2) {
            acc[nt] = __builtin_amdgcn_wmma_f32_16x16x32_bf16(
                false, A, false, B0, (short)0, acc[nt], false, false);
            if (nt + 2 < 16) B0 = *(const v16bf*)(bl + (nt + 2) * 512);
            acc[nt + 1] = __builtin_amdgcn_wmma_f32_16x16x32_bf16(
                false, A, false, B1, (short)0, acc[nt + 1], false, false);
            if (nt + 3 < 16) B1 = *(const v16bf*)(bl + (nt + 3) * 512);
        }
    }

    // Epilogue.  C/D layout: VGPR r, lane L -> N = L%16, M = r + (L/16)*8.
    const int row0 = m0 + wave * 16;
    const int nlo  = lane & 15;
    const int rofs = (lane >> 4) * 8;
    float s = 0.0f;
    if (mode != 0) s = scal[0];
#pragma unroll
    for (int nt = 0; nt < 16; ++nt) {
#pragma unroll
        for (int r = 0; r < 8; ++r) {
            size_t off = (size_t)(row0 + rofs + r) * NHID + nt * 16 + nlo;
            float val = acc[nt][r];
            if (mode == 0) {
                outp[off] = val;
            } else {
                float w = val * up[off] + (1.0f - val) * vp[off];
                outp[off] = s * w + (1.0f - s) * idp[off];
            }
        }
    }
}

// ---------------------------------------------------------------------------
// Final layer: N_OUT = 1 — plain per-row Chebyshev dot product.
// ---------------------------------------------------------------------------
__global__ __launch_bounds__(256) void kan_final(const float* __restrict__ h,
                                                 const float* __restrict__ Cf,
                                                 float* __restrict__ out) {
    int b = blockIdx.x * blockDim.x + threadIdx.x;
    const float* hr = h + (size_t)b * NHID;
    float acc = 0.0f;
    for (int i = 0; i < NHID; ++i) {
        float t = tanhf(hr[i]);
        const float* c = Cf + i * DDEG;
        float p0 = 1.0f, p1 = t;
        acc += c[0] + c[1] * t;
#pragma unroll
        for (int d = 2; d < 8; ++d) {
            float nn = 2.0f * t * p1 - p0; p0 = p1; p1 = nn;
            acc += c[d] * nn;
        }
    }
    out[b] = acc;
}

// ---------------------------------------------------------------------------
extern "C" void kernel_launch(void* const* d_in, const int* in_sizes, int n_in,
                              void* d_out, int out_size, void* d_ws, size_t ws_size,
                              hipStream_t stream) {
    (void)in_sizes; (void)n_in; (void)out_size; (void)ws_size;
    const float* x      = (const float*)d_in[0];
    const float* Brff   = (const float*)d_in[1];
    const float* C_U    = (const float*)d_in[2];
    const float* C_V    = (const float*)d_in[3];
    const float* C_in   = (const float*)d_in[4];   // (4, 256, 256, 8)
    const float* C_out  = (const float*)d_in[5];   // (4, 256, 256, 8)
    const float* alphas = (const float*)d_in[6];
    const float* betas  = (const float*)d_in[7];
    const float* C_f    = (const float*)d_in[8];
    float* out = (float*)d_out;

    // Workspace carve-up (~145 MB total)
    char* p = (char*)d_ws;
    unsigned short* wf = (unsigned short*)p; p += (size_t)10 * WELEMS * sizeof(unsigned short);
    float* h  = (float*)p; p += (size_t)BATCH * NHID * sizeof(float);
    float* u  = (float*)p; p += (size_t)BATCH * NHID * sizeof(float);
    float* v  = (float*)p; p += (size_t)BATCH * NHID * sizeof(float);
    float* t1 = (float*)p; p += (size_t)BATCH * NHID * sizeof(float);

    // 1) Repack all 10 weight tensors to bf16 WMMA fragments
    prep_w<<<128, 256, 0, stream>>>(C_U, wf + (size_t)0 * WELEMS);
    prep_w<<<128, 256, 0, stream>>>(C_V, wf + (size_t)1 * WELEMS);
    for (int bI = 0; bI < 4; ++bI) {
        prep_w<<<128, 256, 0, stream>>>(C_in  + (size_t)bI * WELEMS, wf + (size_t)(2 + bI) * WELEMS);
        prep_w<<<128, 256, 0, stream>>>(C_out + (size_t)bI * WELEMS, wf + (size_t)(6 + bI) * WELEMS);
    }

    // 2) RFF embedding
    rff_kernel<<<(BATCH * 128) / 256, 256, 0, stream>>>(x, Brff, h);

    const int    gblocks = BATCH / MTILE;               // 512
    const size_t shmem   = SMEM_TOTAL;                  // 82944 B -> 3 WGs/WGP

    // 3) u, v projections
    kan_gemm<<<gblocks, 128, shmem, stream>>>(h, wf + (size_t)0 * WELEMS, u,
                                              nullptr, nullptr, nullptr, nullptr, 0);
    kan_gemm<<<gblocks, 128, shmem, stream>>>(h, wf + (size_t)1 * WELEMS, v,
                                              nullptr, nullptr, nullptr, nullptr, 0);

    // 4) Residual gated blocks
    for (int bI = 0; bI < 4; ++bI) {
        kan_gemm<<<gblocks, 128, shmem, stream>>>(h,  wf + (size_t)(2 + bI) * WELEMS, t1,
                                                  u, v, h, betas + bI, 1);
        kan_gemm<<<gblocks, 128, shmem, stream>>>(t1, wf + (size_t)(6 + bI) * WELEMS, h,
                                                  u, v, h, alphas + bI, 1);
    }

    // 5) Final N=1 projection
    kan_final<<<BATCH / 256, 256, 0, stream>>>(h, C_f, out);
}
